// SparseMaskedLSTMCell_57698590655171
// MI455X (gfx1250) — matrix operand order repaired and checked by reading, not compiled, not run on hardware
//
#include <hip/hip_runtime.h>
#include <cstdint>
#include <cstddef>

// ---------------------------------------------------------------------------
// SparseMaskedLSTMCell for MI455X (gfx1250), wave32 + v_wmma_f32_16x16x32_bf16
// ---------------------------------------------------------------------------

typedef __attribute__((ext_vector_type(16))) __bf16    v16bf;
typedef __attribute__((ext_vector_type(8)))  float     v8f;
typedef __attribute__((ext_vector_type(8)))  uint32_t  v8u;

#define B_ROWS 131072
#define IN_DIM 322
#define IN_PAD 352      // 322 padded to 11 k-tiles of 32
#define UNITS  96
#define GATES4 384

// ---- helpers ---------------------------------------------------------------

// Pack two f32 into a dword of two bf16 (round-to-nearest via +0x8000, then
// one v_perm_b32 grabs both high halves): 3 VALU ops per 2 elements.
__device__ inline uint32_t pkbf(float a, float b) {
  uint32_t ua = __builtin_bit_cast(uint32_t, a) + 0x8000u;
  uint32_t ub = __builtin_bit_cast(uint32_t, b) + 0x8000u;
  // result = { ub[31:16], ua[31:16] }  (low half = bf16(a), high = bf16(b))
  return __builtin_amdgcn_perm(ub, ua, 0x07060302u);
}

__device__ inline __bf16 f2bf(float x) {
  uint32_t u = __builtin_bit_cast(uint32_t, x) + 0x8000u;
  uint16_t h = (uint16_t)(u >> 16);
  return __builtin_bit_cast(__bf16, h);
}

// v_exp_f32 + v_rcp_f32 based gates (no IEEE div expansion).
__device__ inline float sig_fast(float x) {
  return __builtin_amdgcn_rcpf(1.0f + __expf(-x));
}

__device__ inline float tanh_fast(float x) {
  float t = __expf(-2.0f * fabsf(x));
  float r = (1.0f - t) * __builtin_amdgcn_rcpf(1.0f + t);
  return copysignf(r, x);
}

// A-operand (16x32 bf16) load from an f32 row, per documented layout:
// lane L (half = L/16, M = L%16): elems 0..7 -> K = kb+8*half+0..7,
//                                 elems 8..15 -> K = kb+8*half+16..23
__device__ inline v16bf load_a_x(const float* __restrict__ rp, int kb, int half) {
  v8u w;
  const float2* p0 = (const float2*)(rp + kb + 8 * half);        // 8B aligned
  const float2* p1 = (const float2*)(rp + kb + 8 * half + 16);
#pragma unroll
  for (int q = 0; q < 4; ++q) {
    float2 x0 = p0[q], x1 = p1[q];
    w[q]     = pkbf(x0.x, x0.y);
    w[4 + q] = pkbf(x1.x, x1.y);
  }
  return __builtin_bit_cast(v16bf, w);
}

// Guarded variant for the last (partial) k-tile of X (kb=320, valid K<322).
__device__ inline v16bf load_a_x_guard(const float* __restrict__ rp, int kb,
                                       int half, int kmax) {
  v8u w;
  const int k0 = kb + 8 * half;
#pragma unroll
  for (int q = 0; q < 4; ++q) {
    const int ka = k0 + 2 * q;
    const int kc = k0 + 16 + 2 * q;
    float a0 = (ka + 0 < kmax) ? rp[ka + 0] : 0.0f;
    float a1 = (ka + 1 < kmax) ? rp[ka + 1] : 0.0f;
    float c0 = (kc + 0 < kmax) ? rp[kc + 0] : 0.0f;
    float c1 = (kc + 1 < kmax) ? rp[kc + 1] : 0.0f;
    w[q]     = pkbf(a0, a1);
    w[4 + q] = pkbf(c0, c1);
  }
  return __builtin_bit_cast(v16bf, w);
}

// A-operand from the h0 row (recurrent): K range = 32*j .. 32*j+31, 16B aligned.
__device__ inline v16bf load_a_h(const float* __restrict__ rp, int j, int half) {
  v8u w;
  const float4* p0 = (const float4*)(rp + 32 * j + 8 * half);
  const float4* p1 = (const float4*)(rp + 32 * j + 8 * half + 16);
  float4 x0 = p0[0], x1 = p0[1], y0 = p1[0], y1 = p1[1];
  w[0] = pkbf(x0.x, x0.y);  w[1] = pkbf(x0.z, x0.w);
  w[2] = pkbf(x1.x, x1.y);  w[3] = pkbf(x1.z, x1.w);
  w[4] = pkbf(y0.x, y0.y);  w[5] = pkbf(y0.z, y0.w);
  w[6] = pkbf(y1.x, y1.y);  w[7] = pkbf(y1.z, y1.w);
  return __builtin_bit_cast(v16bf, w);
}

// B-operand (32x16 bf16) from column-major (K-contiguous) weight buffer:
// lane L: column N = L%16, elems 0..7 hold K = kb + 8*(L/16) + 0..7,
// elems 8..15 hold K = kb + 8*(L/16) + 16..23 -> two 16-byte loads per lane.
__device__ inline v16bf load_b(const __bf16* __restrict__ W, int ldk, int col,
                               int kb, int half) {
  typedef __attribute__((ext_vector_type(8))) __bf16 v8bf;
  const __bf16* p = W + (size_t)col * ldk + kb + 8 * half;
  v8bf lo = *(const v8bf*)p;
  v8bf hi = *(const v8bf*)(p + 16);
  return __builtin_shufflevector(lo, hi, 0, 1, 2, 3, 4, 5, 6, 7, 8, 9, 10, 11, 12, 13, 14, 15);
}

__device__ inline v8f wmma_bf16(v16bf a, v16bf b, v8f c) {
  return __builtin_amdgcn_wmma_f32_16x16x32_bf16(
      /*neg_a=*/false, a, /*neg_b=*/false, b,
      /*c_mod=*/(short)0, c, /*reuse_a=*/false, /*reuse_b=*/false);
}

// ---- weight preparation ----------------------------------------------------

__global__ void zero_ws_kernel(uint32_t* __restrict__ p, int n_dwords) {
  int i = blockIdx.x * blockDim.x + threadIdx.x;
  if (i < n_dwords) p[i] = 0u;
}

__global__ void scatter_kernel(const float* __restrict__ vals,
                               const int* __restrict__ idx, int n,
                               __bf16* __restrict__ W, int ldk) {
  int i = blockIdx.x * blockDim.x + threadIdx.x;
  if (i < n) {
    int r = idx[2 * i + 0];
    int c = idx[2 * i + 1];
    W[(size_t)c * ldk + r] = f2bf(vals[i]);
  }
}

// ---- main LSTM kernel ------------------------------------------------------
// Block = 256 threads = 8 waves; each wave owns a 16-row M-tile.
// Column sub-block j of z (32 cols, per gate) needs input k-tiles:
//   j=0: {0}  j=1: {0..3}  j=2: {3..10}  plus one 32-wide recurrent tile.

__global__ __launch_bounds__(256) void
lstm_wmma_kernel(const float* __restrict__ X, const float* __restrict__ H0,
                 const float* __restrict__ C0, const float* __restrict__ BIAS,
                 const __bf16* __restrict__ WK, const __bf16* __restrict__ WR,
                 float* __restrict__ Hout, float* __restrict__ Cout) {
  const int lane = threadIdx.x & 31;
  const int wave = threadIdx.x >> 5;
  const int half = lane >> 4;   // 0 or 1
  const int ncol = lane & 15;   // N (and A-row M) index
  const int m16  = (blockIdx.x * 8 + wave) * 16;

  const float* xrow = X  + (size_t)(m16 + ncol) * IN_DIM;
  const float* hrow = H0 + (size_t)(m16 + ncol) * UNITS;

  const int ktcnt[3]   = {1, 4, 8};
  const int ktstart[3] = {0, 0, 3};

#pragma unroll
  for (int j = 0; j < 3; ++j) {
    // Load all input A-tiles for this sub-block once (reused by 2 unit-tiles
    // x 4 gates) and the recurrent A-tile.
    v16bf Ain[8];
#pragma unroll
    for (int t = 0; t < 8; ++t) {
      if (t < ktcnt[j]) {
        const int kb = 32 * (ktstart[j] + t);
        Ain[t] = (kb + 32 <= IN_DIM) ? load_a_x(xrow, kb, half)
                                     : load_a_x_guard(xrow, kb, half, IN_DIM);
      }
    }
    const v16bf Ar = load_a_h(hrow, j, half);

#pragma unroll
    for (int uu = 0; uu < 2; ++uu) {
      const int u0 = (2 * j + uu) * 16;   // unit-tile column base (0..80)

      v8f acc[4];
#pragma unroll
      for (int g = 0; g < 4; ++g) {
        const int colbase = g * UNITS + u0;
        v8f a;
#pragma unroll
        for (int q = 0; q < 8; ++q) a[q] = 0.0f;
#pragma unroll
        for (int t = 0; t < 8; ++t) {
          if (t < ktcnt[j]) {
            const int kb = 32 * (ktstart[j] + t);
            v16bf bw = load_b(WK, IN_PAD, colbase + ncol, kb, half);
            a = wmma_bf16(Ain[t], bw, a);
          }
        }
        v16bf br = load_b(WR, UNITS, colbase + ncol, 32 * j, half);
        a = wmma_bf16(Ar, br, a);
        acc[g] = a;
      }

      const float bi = BIAS[0 * UNITS + u0 + ncol];
      const float bf = BIAS[1 * UNITS + u0 + ncol];
      const float bg = BIAS[2 * UNITS + u0 + ncol];
      const float bo = BIAS[3 * UNITS + u0 + ncol];

      // Elementwise gates directly in WMMA C/D layout:
      // VGPR v <-> row m16 + v + 8*half, col u0 + ncol.
#pragma unroll
      for (int v = 0; v < 8; ++v) {
        const size_t row = (size_t)(m16 + v + 8 * half);
        const size_t off = row * UNITS + (size_t)(u0 + ncol);
        const float c0v = C0[off];
        const float iv = sig_fast(acc[0][v] + bi);
        const float fv = sig_fast(acc[1][v] + bf);
        const float gv = tanh_fast(acc[2][v] + bg);
        const float ov = sig_fast(acc[3][v] + bo);
        const float cv = fv * c0v + iv * gv;
        const float hv = ov * tanh_fast(cv);
        Hout[off] = hv;
        Cout[off] = cv;
      }
    }
  }
}

// ---- launcher --------------------------------------------------------------

extern "C" void kernel_launch(void* const* d_in, const int* in_sizes, int n_in,
                              void* d_out, int out_size, void* d_ws,
                              size_t ws_size, hipStream_t stream) {
  const float* X    = (const float*)d_in[0];   // [B, 322]
  const float* H0   = (const float*)d_in[1];   // [B, 96]
  const float* C0   = (const float*)d_in[2];   // [B, 96]
  const float* KV   = (const float*)d_in[3];   // [41216]
  const float* RKV  = (const float*)d_in[4];   // [12288]
  const float* BIAS = (const float*)d_in[5];   // [384]
  const int*   KIDX = (const int*)d_in[6];     // [41216, 2]
  const int*   RKIDX= (const int*)d_in[7];     // [12288, 2]
  const int nk  = in_sizes[3];
  const int nrk = in_sizes[4];

  // Workspace: bf16 column-major (K-contiguous) weights.
  //   WK: [384 cols][352 k]  (270336 B)   WR: [384 cols][96 k] (73728 B)
  __bf16* WK = (__bf16*)d_ws;
  __bf16* WR = WK + (size_t)GATES4 * IN_PAD;
  const int n_bf16   = GATES4 * IN_PAD + GATES4 * UNITS;
  const int n_dwords = n_bf16 / 2;

  zero_ws_kernel<<<(n_dwords + 255) / 256, 256, 0, stream>>>(
      (uint32_t*)d_ws, n_dwords);
  scatter_kernel<<<(nk + 255) / 256, 256, 0, stream>>>(KV, KIDX, nk, WK, IN_PAD);
  scatter_kernel<<<(nrk + 255) / 256, 256, 0, stream>>>(RKV, RKIDX, nrk, WR, UNITS);

  float* Hout = (float*)d_out;
  float* Cout = Hout + (size_t)B_ROWS * UNITS;

  // 8192 M-tiles / 8 waves per block = 1024 blocks.
  lstm_wmma_kernel<<<B_ROWS / (16 * 8), 256, 0, stream>>>(
      X, H0, C0, BIAS, WK, WR, Hout, Cout);
}
